// CATs_24713241822174
// MI455X (gfx1250) — matrix operand order repaired and checked by reading, not compiled, not run on hardware
//
#include <hip/hip_runtime.h>
#include <math.h>

typedef __attribute__((ext_vector_type(16))) _Float16 v16h;
typedef __attribute__((ext_vector_type(8)))  float    v8f;

#define GF_ACCUM  1
#define GF_RELU   2
#define GF_GELU   4
#define GF_TRANSB 8

// param order inside each level (28 tensors per level, base index 9)
enum {
  IP_ec_w1=0, IP_ec_b1, IP_ec_w2, IP_ec_b2, IP_gn_g, IP_gn_b, IP_aff_w, IP_aff_b,
  IP_emb_w1, IP_emb_b1, IP_emb_w2, IP_emb_b2, IP_pin_w, IP_pin_b, IP_ln1_g, IP_ln1_b,
  IP_qkv_w, IP_qkv_b, IP_ao_w, IP_ao_b, IP_ln2_g, IP_ln2_b, IP_mlp_w1, IP_mlp_b1,
  IP_mlp_w2, IP_mlp_b2, IP_pout_w, IP_pout_b
};

// ---------------------------------------------------------------- reductions
__device__ __forceinline__ float blockReduceSum(float v, float* sh) {
  int t = threadIdx.x;
  sh[t] = v; __syncthreads();
  for (int s = 128; s > 0; s >>= 1) { if (t < s) sh[t] += sh[t + s]; __syncthreads(); }
  float r = sh[0]; __syncthreads();
  return r;
}

// ---------------------------------------------------------------- WMMA GEMM
// C[M,N] = act(alpha * A[M,K] x B[K,N] + bias) (+= C when GF_ACCUM)
// A row-major f32 (lda), B row-major f32 (ldb) or [N,K] when GF_TRANSB.
// Block = 4 waves; each wave owns one 16x16 tile (4 M-tiles share one N-tile).
// B tile staged in LDS as f16; A fragment loaded as two contiguous 8-float runs.
// M % 64 == 0, N % 16 == 0, K % 32 == 0 (holds for every GEMM in this net).
__global__ void k_gemm_wmma(const float* __restrict__ A, int lda,
                            const float* __restrict__ B, int ldb,
                            const float* __restrict__ bias,
                            float* __restrict__ C, int ldc,
                            int M, int Nn, int K, float alpha, int flags) {
  __shared__ _Float16 bs[32 * 16];
  const int lane = threadIdx.x, wv = threadIdx.y, tid = wv * 32 + lane;
  const int n0 = blockIdx.x * 16;
  const int m0 = (blockIdx.y * 4 + wv) * 16;
  const int laneHi = lane >> 4, lmod = lane & 15;
  const float* Arow = A + (size_t)(m0 + lmod) * lda;
  v8f acc = {};
  for (int k0 = 0; k0 < K; k0 += 32) {
    // cooperative stage of B[k0:k0+32, n0:n0+16] (shared by all 4 waves)
    #pragma unroll
    for (int j = 0; j < 4; ++j) {
      int e = tid + j * 128;           // e = k*16 + n
      int k = e >> 4, nn = e & 15;
      float v = (flags & GF_TRANSB) ? B[(size_t)(n0 + nn) * ldb + k0 + k]
                                    : B[(size_t)(k0 + k) * ldb + n0 + nn];
      bs[e] = (_Float16)v;
    }
    __syncthreads();
    // B fragment: lane holds column n=lmod, K = laneHi*16 + e
    v16h bf;
    #pragma unroll
    for (int e = 0; e < 16; ++e) bf[e] = bs[(laneHi * 16 + e) * 16 + lmod];
    // A fragment: lane holds row m, K = laneHi*16 + e
    const float* a0 = Arow + k0 + laneHi * 16;
    if (k0 + 32 < K) __builtin_prefetch(a0 + 32, 0, 1);
    v16h af;
    #pragma unroll
    for (int e = 0; e < 8; ++e) af[e] = (_Float16)a0[e];
    #pragma unroll
    for (int e = 0; e < 8; ++e) af[8 + e] = (_Float16)a0[8 + e];
    acc = __builtin_amdgcn_wmma_f32_16x16x32_f16(false, af, false, bf,
                                                 (short)0, acc, false, false);
    __syncthreads();
  }
  // D layout: element r -> (m = laneHi*8 + r, n = lmod)
  #pragma unroll
  for (int r = 0; r < 8; ++r) {
    int mm = m0 + laneHi * 8 + r, nn = n0 + lmod;
    float v = acc[r] * alpha;
    if (bias) v += bias[nn];
    if (flags & GF_RELU) v = fmaxf(v, 0.0f);
    if (flags & GF_GELU) {
      float x = v;
      v = 0.5f * x * (1.0f + tanhf(0.7978845608f * (x + 0.044715f * x * x * x)));
    }
    size_t off = (size_t)mm * ldc + nn;
    if (flags & GF_ACCUM) v += C[off];
    C[off] = v;
  }
}

// ------------------------------------------------- implicit-GEMM 3x3/s2 conv
// y[oc, oy, ox] = relu(bias[oc] + sum_{ic,kh,kw} w[oc,ic,kh,kw]*x[ic,iy,ix])
// stride 2, pad 1; OC = 128, IC % 32 == 0, OH*OW % 64 == 0.
__global__ void k_conv2d_relu_wmma(const float* __restrict__ x,
                                   const float* __restrict__ w,
                                   const float* __restrict__ bias,
                                   float* __restrict__ y,
                                   int IC, int IH, int IW, int OH, int OW) {
  __shared__ _Float16 bs[32 * 16];
  const int lane = threadIdx.x, wv = threadIdx.y, tid = wv * 32 + lane;
  const int n0 = blockIdx.x * 16;                 // output-channel tile
  const int m0 = (blockIdx.y * 4 + wv) * 16;      // spatial tile
  const int laneHi = lane >> 4, lmod = lane & 15;
  const int M = OH * OW;
  const int p = m0 + lmod;
  const int oy = p / OW, ox = p % OW;
  v8f acc = {};
  for (int kh = 0; kh < 3; ++kh) {
    const int iy = oy * 2 - 1 + kh;
    for (int kw = 0; kw < 3; ++kw) {
      const int ix = ox * 2 - 1 + kw;
      const bool inb = (iy >= 0 && iy < IH && ix >= 0 && ix < IW);
      for (int icb = 0; icb < IC; icb += 32) {
        #pragma unroll
        for (int j = 0; j < 4; ++j) {
          int e = tid + j * 128;
          int k = e >> 4, nn = e & 15;
          bs[e] = (_Float16)w[((size_t)(n0 + nn) * IC + icb + k) * 9 + kh * 3 + kw];
        }
        __syncthreads();
        v16h bf;
        #pragma unroll
        for (int e = 0; e < 16; ++e) bf[e] = bs[(laneHi * 16 + e) * 16 + lmod];
        v16h af;
        #pragma unroll
        for (int e = 0; e < 16; ++e) {
          int ic = icb + laneHi * 16 + e;
          float v = inb ? x[((size_t)ic * IH + iy) * IW + ix] : 0.0f;
          af[e] = (_Float16)v;
        }
        acc = __builtin_amdgcn_wmma_f32_16x16x32_f16(false, af, false, bf,
                                                     (short)0, acc, false, false);
        __syncthreads();
      }
    }
  }
  #pragma unroll
  for (int r = 0; r < 8; ++r) {
    int mm = m0 + laneHi * 8 + r, nn = n0 + lmod;
    y[(size_t)nn * M + mm] = fmaxf(acc[r] + bias[nn], 0.0f);
  }
}

// ----------------------------------------------------------- direct conv4d
// out1: conv over (Ht,Wt) of x subsampled (Hs,Ws); out2: conv over (Hs,Ws)
// of x subsampled (Ht,Wt).  y shape (O,oHt,oWt,oHs,oWs).
__global__ void k_conv4d(const float* __restrict__ x,
                         const float* __restrict__ w1, const float* __restrict__ b1,
                         const float* __restrict__ w2, const float* __restrict__ b2,
                         float* __restrict__ y,
                         int C, int O, int Ht, int Wt, int Hs, int Ws,
                         int oHt, int oWt, int oHs, int oWs,
                         int st, int ss, int pt, int ps, int kt) {
  size_t total = (size_t)O * oHt * oWt * oHs * oWs;
  size_t i = (size_t)blockIdx.x * blockDim.x + threadIdx.x;
  if (i >= total) return;
  int ows = i % oWs; size_t t = i / oWs;
  int ohs = t % oHs; t /= oHs;
  int owt = t % oWt; t /= oWt;
  int oht = t % oHt; int o = (int)(t / oHt);
  const int HsWs = Hs * Ws;
  float acc = b1[o] + b2[o];
  for (int c = 0; c < C; ++c) {
    const float* xc = x + (size_t)c * Ht * Wt * HsWs;
    // term 1: subsample s-dims, convolve t-dims
    int hs1 = ohs * ss, ws1 = ows * ss;
    const float* w1o = w1 + ((size_t)o * C + c) * kt * kt;
    for (int kh = 0; kh < kt; ++kh) {
      int hti = oht * st - pt + kh; if (hti < 0 || hti >= Ht) continue;
      for (int kw = 0; kw < kt; ++kw) {
        int wti = owt * st - pt + kw; if (wti < 0 || wti >= Wt) continue;
        acc += w1o[kh * kt + kw] * xc[(size_t)(hti * Wt + wti) * HsWs + hs1 * Ws + ws1];
      }
    }
    // term 2: subsample t-dims, convolve s-dims (3x3)
    const float* xc2 = xc + (size_t)(oht * st * Wt + owt * st) * HsWs;
    const float* w2o = w2 + ((size_t)o * C + c) * 9;
    for (int kh = 0; kh < 3; ++kh) {
      int hsi = ohs * ss - ps + kh; if (hsi < 0 || hsi >= Hs) continue;
      for (int kw = 0; kw < 3; ++kw) {
        int wsi = ows * ss - ps + kw; if (wsi < 0 || wsi >= Ws) continue;
        acc += w2o[kh * 3 + kw] * xc2[hsi * Ws + wsi];
      }
    }
  }
  y[i] = acc;
}

// ----------------------------------------------------------- group norm (G=1)
__global__ void k_red_partial(const float* __restrict__ x, int n, float* __restrict__ part) {
  __shared__ float sh[256];
  float s = 0.f, s2 = 0.f;
  for (int i = blockIdx.x * 256 + threadIdx.x; i < n; i += gridDim.x * 256) {
    float v = x[i]; s += v; s2 += v * v;
  }
  s = blockReduceSum(s, sh);
  s2 = blockReduceSum(s2, sh);
  if (threadIdx.x == 0) { part[2 * blockIdx.x] = s; part[2 * blockIdx.x + 1] = s2; }
}
__global__ void k_red_final(const float* __restrict__ part, int nb, float inv_n,
                            float* __restrict__ stats) {
  __shared__ float sh[256];
  float s = 0.f, s2 = 0.f;
  for (int i = threadIdx.x; i < nb; i += 256) { s += part[2 * i]; s2 += part[2 * i + 1]; }
  s = blockReduceSum(s, sh);
  s2 = blockReduceSum(s2, sh);
  if (threadIdx.x == 0) {
    float mu = s * inv_n;
    float var = s2 * inv_n - mu * mu;
    stats[0] = mu; stats[1] = rsqrtf(var + 1e-5f);
  }
}
__global__ void k_gn_apply_relu(float* __restrict__ y, const float* __restrict__ stats,
                                const float* __restrict__ g, const float* __restrict__ b,
                                int chanS, int n) {
  int i = blockIdx.x * 256 + threadIdx.x;
  if (i >= n) return;
  int c = i / chanS;
  float v = (y[i] - stats[0]) * stats[1] * g[c] + b[c];
  y[i] = fmaxf(v, 0.0f);
}

// ----------------------------------------------------------------- layernorm
__global__ void k_layernorm(const float* __restrict__ x, const float* __restrict__ g,
                            const float* __restrict__ b, float* __restrict__ y, int D) {
  __shared__ float sh[256];
  const float* row = x + (size_t)blockIdx.x * D;
  float* orow = y + (size_t)blockIdx.x * D;
  float s = 0.f, s2 = 0.f;
  for (int d = threadIdx.x; d < D; d += 256) { float v = row[d]; s += v; s2 += v * v; }
  s = blockReduceSum(s, sh);
  s2 = blockReduceSum(s2, sh);
  float mu = s / D, var = s2 / D - mu * mu, r = rsqrtf(var + 1e-5f);
  for (int d = threadIdx.x; d < D; d += 256) orow[d] = (row[d] - mu) * r * g[d] + b[d];
}

// --------------------------------------------------------------- row softmax
__global__ void k_softmax_rows(float* __restrict__ S, int N) {
  __shared__ float sh[256];
  float* row = S + ((size_t)blockIdx.y * N + blockIdx.x) * N;
  float m = -1e30f;
  for (int j = threadIdx.x; j < N; j += 256) m = fmaxf(m, row[j]);
  sh[threadIdx.x] = m; __syncthreads();
  for (int s = 128; s > 0; s >>= 1) {
    if (threadIdx.x < s) sh[threadIdx.x] = fmaxf(sh[threadIdx.x], sh[threadIdx.x + s]);
    __syncthreads();
  }
  m = sh[0]; __syncthreads();
  float sum = 0.f;
  for (int j = threadIdx.x; j < N; j += 256) { float e = __expf(row[j] - m); row[j] = e; sum += e; }
  sum = blockReduceSum(sum, sh);
  float inv = 1.0f / sum;
  for (int j = threadIdx.x; j < N; j += 256) row[j] *= inv;
}

// --------------------------------------------------------- permutes / packs
__global__ void k_transpose4d(const float* __restrict__ x, float* __restrict__ y, int S) {
  size_t n = (size_t)4 * S * S * S * S;
  size_t i = (size_t)blockIdx.x * 256 + threadIdx.x;
  if (i >= n) return;
  int b = i % S; size_t t = i / S;
  int a = t % S; t /= S;
  int e = t % S; t /= S;
  int d = t % S; int c = (int)(t / S);
  y[i] = x[((((size_t)c * S + a) * S + b) * S + d) * S + e];
}
__global__ void k_pack_x(const float* __restrict__ e, float* __restrict__ xt, int N) {
  size_t n = (size_t)N * 256;
  size_t i = (size_t)blockIdx.x * 256 + threadIdx.x;
  if (i >= n) return;
  int f = (int)(i % 256); int tok = (int)(i / 256);
  xt[i] = e[(size_t)f * N + tok];
}
__global__ void k_pack_a(const float* __restrict__ feat, float* __restrict__ z, int N) {
  size_t n = (size_t)N * 128;
  size_t i = (size_t)blockIdx.x * 256 + threadIdx.x;
  if (i >= n) return;
  int c = (int)(i % 128); int tok = (int)(i / 128);
  z[(size_t)tok * 640 + 512 + c] = feat[(size_t)c * N + tok];
}
__global__ void k_unpack_y(const float* __restrict__ yt, float* __restrict__ y6, int N) {
  size_t n = (size_t)256 * N;
  size_t i = (size_t)blockIdx.x * 256 + threadIdx.x;
  if (i >= n) return;
  int s = (int)(i % N); int f = (int)(i / N);
  y6[i] = yt[(size_t)s * 256 + f];
}

// ----------------------------------------------------------- interp4d (+add)
// x (C,i0,i1,i2,i3) -> bilinear^4 at (C,o0,o1,o2,o3).  mode: 0=write, 1=add,
// 2=add of the 4d-transposed result (dst(c,a,b,d,e) += res(c,d,e,a,b)).
__global__ void k_interp4d(const float* __restrict__ x, float* __restrict__ y, int C,
                           int i0, int i1, int i2, int i3,
                           int o0, int o1, int o2, int o3, int mode) {
  size_t n = (size_t)C * o0 * o1 * o2 * o3;
  size_t idx = (size_t)blockIdx.x * 256 + threadIdx.x;
  if (idx >= n) return;
  int e = idx % o3; size_t t = idx / o3;
  int d = t % o2; t /= o2;
  int b = t % o1; t /= o1;
  int a = t % o0; int c = (int)(t / o0);
  int qq[4] = {a, b, d, e};
  if (mode == 2) { qq[0] = d; qq[1] = e; qq[2] = a; qq[3] = b; }
  int ii[4] = {i0, i1, i2, i3};
  int oo[4] = {o0, o1, o2, o3};
  int lo[4], hi[4]; float w[4];
  #pragma unroll
  for (int ax = 0; ax < 4; ++ax) {
    if (oo[ax] == ii[ax]) { lo[ax] = qq[ax]; hi[ax] = qq[ax]; w[ax] = 0.f; }
    else {
      float pos = (oo[ax] > 1) ? qq[ax] * (float)(ii[ax] - 1) / (float)(oo[ax] - 1) : 0.f;
      int l = (int)floorf(pos);
      hi[ax] = (l + 1 < ii[ax]) ? l + 1 : ii[ax] - 1;
      lo[ax] = l; w[ax] = pos - (float)l;
    }
  }
  const float* xc = x + (size_t)c * i0 * i1 * i2 * i3;
  float acc = 0.f;
  #pragma unroll
  for (int m = 0; m < 16; ++m) {
    int s0 = (m & 8) ? hi[0] : lo[0];
    int s1 = (m & 4) ? hi[1] : lo[1];
    int s2 = (m & 2) ? hi[2] : lo[2];
    int s3 = (m & 1) ? hi[3] : lo[3];
    float wt = ((m & 8) ? w[0] : 1.f - w[0]) * ((m & 4) ? w[1] : 1.f - w[1]) *
               ((m & 2) ? w[2] : 1.f - w[2]) * ((m & 1) ? w[3] : 1.f - w[3]);
    if (wt != 0.f)
      acc += wt * xc[(((size_t)s0 * i1 + s1) * i2 + s2) * i3 + s3];
  }
  if (mode == 0) y[idx] = acc; else y[idx] += acc;
}

// -------------------------------------------------------------- final flow
__global__ void k_flow(const float* __restrict__ c3, float* __restrict__ out, int S) {
  __shared__ float sh[256];
  const int ht = blockIdx.x / S, wt = blockIdx.x % S;
  const int n = S * S;
  const size_t chs = (size_t)S * S * S * S;
  const float* base = c3 + ((size_t)ht * S + wt) * n;
  const float scale = 0.25f * (1.0f / 0.02f); // mean over 4 ch, / BETA
  float m = -1e30f;
  for (int j = threadIdx.x; j < n; j += 256) {
    float v = 0.f;
    for (int c = 0; c < 4; ++c) v += base[(size_t)c * chs + j];
    m = fmaxf(m, v * scale);
  }
  sh[threadIdx.x] = m; __syncthreads();
  for (int s = 128; s > 0; s >>= 1) {
    if (threadIdx.x < s) sh[threadIdx.x] = fmaxf(sh[threadIdx.x], sh[threadIdx.x + s]);
    __syncthreads();
  }
  m = sh[0]; __syncthreads();
  const float inv2 = 2.0f / (float)(S - 1);
  float se = 0.f, sx = 0.f, sy = 0.f;
  for (int j = threadIdx.x; j < n; j += 256) {
    float v = 0.f;
    for (int c = 0; c < 4; ++c) v += base[(size_t)c * chs + j];
    float e = __expf(v * scale - m);
    int hs = j / S, ws = j % S;
    se += e; sx += e * (-1.0f + ws * inv2); sy += e * (-1.0f + hs * inv2);
  }
  se = blockReduceSum(se, sh);
  sx = blockReduceSum(sx, sh);
  sy = blockReduceSum(sy, sh);
  if (threadIdx.x == 0) {
    float gx = sx / se, gy = sy / se;
    float mx = (gx + 1.0f) * (S - 1) * 0.5f;
    float my = (gy + 1.0f) * (S - 1) * 0.5f;
    out[blockIdx.x] = mx - (float)wt;
    out[(size_t)S * S + blockIdx.x] = my - (float)ht;
  }
}

// ================================================================== host side
static inline unsigned cdivz(size_t a, int b) { return (unsigned)((a + b - 1) / b); }

static void gemm(hipStream_t s, const float* A, int lda, const float* B, int ldb,
                 const float* bias, float* C, int ldc, int M, int Nn, int K,
                 float alpha, int flags) {
  dim3 blk(32, 4), grd(Nn / 16, M / 64);
  k_gemm_wmma<<<grd, blk, 0, s>>>(A, lda, B, ldb, bias, C, ldc, M, Nn, K, alpha, flags);
}

static void gn_relu(hipStream_t s, float* y, size_t n, const float* g, const float* b,
                    float* part, float* stats) {
  unsigned nb = cdivz(n, 256); if (nb > 4096u) nb = 4096u;
  k_red_partial<<<nb, 256, 0, s>>>(y, (int)n, part);
  k_red_final<<<1, 256, 0, s>>>(part, (int)nb, 1.0f / (float)n, stats);
  k_gn_apply_relu<<<cdivz(n, 256), 256, 0, s>>>(y, stats, g, b, (int)(n / 4), (int)n);
}

static void run_transformer(hipStream_t s, void* const* d_in, int L,
                            const float* cin, float* cdst, int addMode,
                            const float* feat,
                            float* embb, float* xtok, float* z, float* hb,
                            float* qkvb, float* Sb, float* ob, float* mb,
                            float* yb, float* y6) {
  auto P = [&](int j) { return (const float*)d_in[9 + L * 28 + j]; };
  static const int stA[3] = {1, 2, 4}, ptA[3] = {1, 1, 2}, ktA[3] = {3, 3, 5};
  const int S = 8 << L, N = S * S;
  // emb = conv4d(c): (4,S,S,S,S) -> (4,8,8,S,S)
  size_t en = (size_t)4 * 8 * 8 * N;
  k_conv4d<<<cdivz(en, 256), 256, 0, s>>>(cin, P(IP_emb_w1), P(IP_emb_b1),
      P(IP_emb_w2), P(IP_emb_b2), embb, 4, 4, S, S, S, S, 8, 8, S, S,
      stA[L], 1, ptA[L], 1, ktA[L]);
  // tokens (N,256)
  k_pack_x<<<cdivz((size_t)N * 256, 256), 256, 0, s>>>(embb, xtok, N);
  // z[:, :512] = x @ pin + b ; z[:, 512:640] = feat^T
  gemm(s, xtok, 256, P(IP_pin_w), 512, P(IP_pin_b), z, 640, N, 512, 256, 1.f, 0);
  k_pack_a<<<cdivz((size_t)N * 128, 256), 256, 0, s>>>(feat, z, N);
  // attention block
  k_layernorm<<<N, 256, 0, s>>>(z, P(IP_ln1_g), P(IP_ln1_b), hb, 640);
  gemm(s, hb, 640, P(IP_qkv_w), 1920, P(IP_qkv_b), qkvb, 1920, N, 1920, 640, 1.f, 0);
  const float sc = 1.0f / sqrtf(160.0f);
  for (int h = 0; h < 4; ++h)   // S_h = Q_h K_h^T / sqrt(hd)
    gemm(s, qkvb + h * 160, 1920, qkvb + 640 + h * 160, 1920, nullptr,
         Sb + (size_t)h * N * N, N, N, N, 160, sc, GF_TRANSB);
  { dim3 g(N, 4); k_softmax_rows<<<g, 256, 0, s>>>(Sb, N); }
  for (int h = 0; h < 4; ++h)   // O_h = P_h V_h
    gemm(s, Sb + (size_t)h * N * N, N, qkvb + 1280 + h * 160, 1920, nullptr,
         ob + h * 160, 640, N, 160, N, 1.f, 0);
  gemm(s, ob, 640, P(IP_ao_w), 640, P(IP_ao_b), z, 640, N, 640, 640, 1.f, GF_ACCUM);
  // MLP block
  k_layernorm<<<N, 256, 0, s>>>(z, P(IP_ln2_g), P(IP_ln2_b), hb, 640);
  gemm(s, hb, 640, P(IP_mlp_w1), 2560, P(IP_mlp_b1), mb, 2560, N, 2560, 640, 1.f, GF_GELU);
  gemm(s, mb, 2560, P(IP_mlp_w2), 640, P(IP_mlp_b2), z, 640, N, 640, 2560, 1.f, GF_ACCUM);
  // project out, unpack, upsample target dims, accumulate into cdst
  gemm(s, z, 640, P(IP_pout_w), 256, P(IP_pout_b), yb, 256, N, 256, 640, 1.f, 0);
  k_unpack_y<<<cdivz((size_t)256 * N, 256), 256, 0, s>>>(yb, y6, N);
  size_t on = (size_t)4 * S * S * S * S;
  k_interp4d<<<cdivz(on, 256), 256, 0, s>>>(y6, cdst, 4, 8, 8, S, S, S, S, S, S, addMode);
}

extern "C" void kernel_launch(void* const* d_in, const int* in_sizes, int n_in,
                              void* d_out, int out_size, void* d_ws, size_t ws_size,
                              hipStream_t stream) {
  (void)in_sizes; (void)n_in; (void)out_size; (void)ws_size;
  const float* corr[3]  = {(const float*)d_in[0], (const float*)d_in[1], (const float*)d_in[2]};
  const float* tfeat[3] = {(const float*)d_in[3], (const float*)d_in[4], (const float*)d_in[5]};
  const float* sfeat[3] = {(const float*)d_in[6], (const float*)d_in[7], (const float*)d_in[8]};
  auto P = [&](int L, int j) { return (const float*)d_in[9 + L * 28 + j]; };

  static const int FEATC[3] = {2048, 1024, 512};
  static const int INCH[3]  = {3, 23, 4};
  static const int SDIM[3]  = {8, 16, 32};
  static const int HIN[3]   = {16, 32, 64};

  // ---- workspace bump allocator (floats) ----
  float* w = (float*)d_ws;
  size_t off = 0;
  auto alloc = [&](size_t nf) { float* p = w + off; off += ((nf + 63) & ~(size_t)63); return p; };
  float *tf[3], *sf[3], *cbuf[3];
  for (int i = 0; i < 3; ++i) {
    tf[i] = alloc((size_t)128 * SDIM[i] * SDIM[i]);
    sf[i] = alloc((size_t)128 * SDIM[i] * SDIM[i]);
  }
  for (int i = 0; i < 3; ++i) cbuf[i] = alloc((size_t)4 * SDIM[i] * SDIM[i] * SDIM[i] * SDIM[i]);
  const int Nmax = 1024, Smax = 32;
  float* ct    = alloc((size_t)4 * Smax * Smax * Smax * Smax);
  float* embb  = alloc((size_t)256 * Nmax);
  float* xtok  = alloc((size_t)Nmax * 256);
  float* z     = alloc((size_t)Nmax * 640);
  float* hb    = alloc((size_t)Nmax * 640);
  float* qkvb  = alloc((size_t)Nmax * 1920);
  float* Sb    = alloc((size_t)4 * Nmax * Nmax);
  float* ob    = alloc((size_t)Nmax * 640);
  float* mb    = Sb;                     // alias: attention scores dead by MLP time
  float* yb    = alloc((size_t)Nmax * 256);
  float* y6    = alloc((size_t)256 * Nmax);
  float* part  = alloc(2 * 4096 + 8);
  float* stats = alloc(8);

  // ---- stage A: aff features (WMMA implicit-GEMM conv, stride 2, relu) ----
  for (int i = 0; i < 3; ++i) {
    size_t lastoff = (size_t)FEATC[i] * HIN[i] * HIN[i];   // [:, -1]
    int OH = HIN[i] / 2, M = OH * OH;
    dim3 blk(32, 4), grd(128 / 16, M / 64);
    k_conv2d_relu_wmma<<<grd, blk, 0, stream>>>(tfeat[i] + lastoff, P(i, IP_aff_w),
        P(i, IP_aff_b), tf[i], FEATC[i], HIN[i], HIN[i], OH, OH);
    k_conv2d_relu_wmma<<<grd, blk, 0, stream>>>(sfeat[i] + lastoff, P(i, IP_aff_w),
        P(i, IP_aff_b), sf[i], FEATC[i], HIN[i], HIN[i], OH, OH);
  }

  // ---- stages B+C per level: early conv4d + GN/relu, chain, aggregate ----
  for (int i = 0; i < 3; ++i) {
    int S = SDIM[i], H = HIN[i];
    size_t outn = (size_t)4 * S * S * S * S;
    k_conv4d<<<cdivz(outn, 256), 256, 0, stream>>>(corr[i], P(i, IP_ec_w1), P(i, IP_ec_b1),
        P(i, IP_ec_w2), P(i, IP_ec_b2), cbuf[i], INCH[i], 4, H, H, H, H, S, S, S, S,
        2, 2, 1, 1, 3);
    gn_relu(stream, cbuf[i], outn, P(i, IP_gn_g), P(i, IP_gn_b), part, stats);
    if (i > 0) {
      int Sp = SDIM[i - 1];
      k_interp4d<<<cdivz(outn, 256), 256, 0, stream>>>(cbuf[i - 1], cbuf[i], 4,
          Sp, Sp, Sp, Sp, S, S, S, S, 1);
    }
    // snapshot transpose before the in-place aggregation
    k_transpose4d<<<cdivz(outn, 256), 256, 0, stream>>>(cbuf[i], ct, S);
    run_transformer(stream, d_in, i, cbuf[i], cbuf[i], 1, sf[i],
                    embb, xtok, z, hb, qkvb, Sb, ob, mb, yb, y6);
    run_transformer(stream, d_in, i, ct, cbuf[i], 2, tf[i],
                    embb, xtok, z, hb, qkvb, Sb, ob, mb, yb, y6);
  }

  // ---- final: channel mean -> softmax over (hs,ws) -> soft-argmax flow ----
  k_flow<<<32 * 32, 256, 0, stream>>>(cbuf[2], (float*)d_out, 32);
}
